// EPTM_32736240730644
// MI455X (gfx1250) — matrix operand order repaired and checked by reading, not compiled, not run on hardware
//
#include <hip/hip_runtime.h>
#include <hip/hip_bf16.h>
#include <math.h>

typedef __attribute__((ext_vector_type(16))) __bf16 v16bf;
typedef __attribute__((ext_vector_type(8)))  __bf16 v8bf;
typedef __attribute__((ext_vector_type(8)))  float  v8f;

#define BATCH 32
#define SLEN  2048
#define DDIM  512
#define HDIM  512
#define MTILE 64            // s-rows per block (4 M-subtiles of 16)
#define LDS_STRIDE 520      // bf16 elems; 1040B rows -> conflict-free b128 A reads
#define CTX_LDS_BYTES (MTILE * LDS_STRIDE * 2)
#define DYN_LDS_BYTES (CTX_LDS_BYTES + MTILE * 4)

// Hardware tanh if the toolchain exposes it (CDNA5 has V_TANH_F32).
#if __has_builtin(__builtin_amdgcn_tanhf)
  #define FAST_TANH(x) __builtin_amdgcn_tanhf(x)
#elif __has_builtin(__builtin_amdgcn_tanh_f32)
  #define FAST_TANH(x) __builtin_amdgcn_tanh_f32(x)
#else
  #define FAST_TANH(x) tanhf(x)
#endif

// ---------------------------------------------------------------------------
// K1: repack Wc0..Wc3 (f32, [H][D]) into bf16 WMMA B-fragments.
// Bfrag[((p*32+ht)*16+ds)*32+lane][j] =
//   Wc_p[ h = ht*16 + (lane&15) ][ d = ds*32 + 16*(lane>=16) + j ]
// ---------------------------------------------------------------------------
__global__ void prep_w(const float* __restrict__ W0, const float* __restrict__ W1,
                       const float* __restrict__ W2, const float* __restrict__ W3,
                       __bf16* __restrict__ Bfrag) {
    int t    = blockIdx.x * 256 + threadIdx.x;   // 0..65535
    int lane = t & 31;
    int ds   = (t >> 5) & 15;
    int ht   = (t >> 9) & 31;
    int p    = (t >> 14) & 3;
    const float* W = (p == 0) ? W0 : (p == 1) ? W1 : (p == 2) ? W2 : W3;
    int h = ht * 16 + (lane & 15);
    int d = ds * 32 + ((lane >> 4) << 4);
    const float* src = W + (size_t)h * DDIM + d;
    __bf16* dst = Bfrag + (size_t)t * 16;
#pragma unroll
    for (int j = 0; j < 16; ++j) dst[j] = (__bf16)src[j];
}

// ---------------------------------------------------------------------------
// K2: inp[b][h] = x[b] . Wi[h] + bi[h]   (32x512x512, tiny)
// ---------------------------------------------------------------------------
__global__ void inp_kernel(const float* __restrict__ x, const float* __restrict__ Wi,
                           const float* __restrict__ bi, float* __restrict__ inp) {
    int t = blockIdx.x * 256 + threadIdx.x;      // 0..16383
    int b = t >> 9, h = t & 511;
    const float4* xv = (const float4*)(x + (size_t)b * HDIM);
    const float4* wv = (const float4*)(Wi + (size_t)h * HDIM);
    float s = bi[h];
#pragma unroll 4
    for (int i = 0; i < HDIM / 4; ++i) {
        float4 a = xv[i], w = wv[i];
        s += a.x * w.x + a.y * w.y + a.z * w.z + a.w * w.w;
    }
    inp[t] = s;
}

// ---------------------------------------------------------------------------
// K3: fused GEMM + tanh + V-dot.
// Block = (b, 64-row s-tile). 8 waves each own a 64-col N-eighth.
// ks is the outer compute loop; 16 live accumulators acc[mt][nt] so that
// each A fragment (LDS) is loaded once per ks and each B fragment (global)
// feeds 4 WMMAs. Nothing is loop-invariant -> no hoist-and-spill.
// ---------------------------------------------------------------------------
__global__ __launch_bounds__(256) void att_kernel(
    const float* __restrict__ context, const __bf16* __restrict__ Bfrag,
    const float* __restrict__ inp,
    const float* __restrict__ bc0, const float* __restrict__ bc1,
    const float* __restrict__ bc2, const float* __restrict__ bc3,
    const float* __restrict__ V, float* __restrict__ att) {

    extern __shared__ char smem_raw[];
    __bf16* ctx_lds = (__bf16*)smem_raw;
    float*  att_lds = (float*)(smem_raw + CTX_LDS_BYTES);

    int blk = blockIdx.x;        // 0..1023
    int b   = blk >> 5;
    int s0  = (blk & 31) * MTILE;
    int tid = threadIdx.x;

    // Stage context tile (f32 -> bf16) into LDS, float4-coalesced.
    const float4* src = (const float4*)(context + ((size_t)b * SLEN + s0) * DDIM);
    for (int i = tid; i < MTILE * DDIM / 4; i += 256) {
        int row = i >> 7;          // 128 float4 per row
        int c4  = i & 127;
        float4 v = src[row * (DDIM / 4) + c4];
        __bf16* dst = &ctx_lds[row * LDS_STRIDE + c4 * 4];
        dst[0] = (__bf16)v.x; dst[1] = (__bf16)v.y;
        dst[2] = (__bf16)v.z; dst[3] = (__bf16)v.w;
    }
    __syncthreads();

    int lane = tid & 31;
    int wave = tid >> 5;                  // 0..7 -> N-eighth (64 h cols)
    int n    = lane & 15;                 // N within 16-tile
    int kOff = (lane < 16) ? 0 : 16;      // contiguous K order (matches Bfrag)
    int arow = lane & 15;                 // A row within M-subtile

#pragma unroll 1
    for (int kp = 0; kp < 4; ++kp) {
        if (tid < MTILE) att_lds[tid] = 0.0f;
        __syncthreads();
        const float* bc = (kp == 0) ? bc0 : (kp == 1) ? bc1 : (kp == 2) ? bc2 : bc3;

        v8f acc[4][4];                    // [mt][nt], 128 VGPRs
#pragma unroll
        for (int mt = 0; mt < 4; ++mt)
#pragma unroll
            for (int nt = 0; nt < 4; ++nt) acc[mt][nt] = (v8f){};

        // base of this wave's B fragments for this projection
        const __bf16* bbase =
            Bfrag + ((((size_t)kp * 32 + wave * 4) * 16) * 32 + lane) * 16;

#pragma unroll 4
        for (int ks = 0; ks < 16; ++ks) {
            // A fragments for the 4 M-subtiles: loaded once per ks
            v16bf a[4];
#pragma unroll
            for (int mt = 0; mt < 4; ++mt) {
                const __bf16* ap =
                    &ctx_lds[(mt * 16 + arow) * LDS_STRIDE + ks * 32 + kOff];
                v8bf a0 = *(const v8bf*)ap;            // K = kOff..kOff+7
                v8bf a1 = *(const v8bf*)(ap + 8);      // K = kOff+8..kOff+15
                a[mt] = __builtin_shufflevector(a0, a1,
                          0,1,2,3,4,5,6,7,8,9,10,11,12,13,14,15);
            }
#pragma unroll
            for (int nt = 0; nt < 4; ++nt) {
                // B fragment (kp, ht = wave*4+nt, ks): 32B/lane, 4x reused
                v16bf bb = *(const v16bf*)(bbase +
                           ((size_t)nt * 16 * 32 + (size_t)ks * 32) * 16);
#pragma unroll
                for (int mt = 0; mt < 4; ++mt) {
                    acc[mt][nt] = __builtin_amdgcn_wmma_f32_16x16x32_bf16(
                                      false, a[mt], false, bb, (short)0,
                                      acc[mt][nt], false, false);
                }
            }
        }

        // epilogue: tanh + V-weighted accumulation.
        // C/D layout: vgpr r, lane l -> M = r + 8*(l>>4), N = l&15
        float accS[4][8];
#pragma unroll
        for (int mt = 0; mt < 4; ++mt)
#pragma unroll
            for (int r = 0; r < 8; ++r) accS[mt][r] = 0.0f;

#pragma unroll
        for (int nt = 0; nt < 4; ++nt) {
            int h  = (wave * 4 + nt) * 16 + n;
            float inb = inp[b * HDIM + h] + bc[h];
            float vv  = V[h];
#pragma unroll
            for (int mt = 0; mt < 4; ++mt)
#pragma unroll
                for (int r = 0; r < 8; ++r)
                    accS[mt][r] += vv * FAST_TANH(inb + acc[mt][nt][r]);
        }

        // reduce over the 16 lanes of each half (sums over N within tile)
#pragma unroll
        for (int mt = 0; mt < 4; ++mt) {
#pragma unroll
            for (int r = 0; r < 8; ++r) {
                float v = accS[mt][r];
                v += __shfl_xor(v, 1, 32);
                v += __shfl_xor(v, 2, 32);
                v += __shfl_xor(v, 4, 32);
                v += __shfl_xor(v, 8, 32);
                if ((lane & 15) == 0) {
                    int m = r + (lane >> 4) * 8;
                    atomicAdd(&att_lds[mt * 16 + m], v);
                }
            }
        }
        __syncthreads();
        if (tid < MTILE)
            att[((size_t)b * 4 + kp) * SLEN + s0 + tid] = att_lds[tid];
        __syncthreads();
    }
}

// ---------------------------------------------------------------------------
// K4: per batch: scores = 10*tanh(att), masked argmax (first occurrence),
// p = 1/sum(exp(score-max)), plus mask passthrough.
// out = [indices(32) | p(32) | mask(32*2048)]
// ---------------------------------------------------------------------------
__global__ void out_kernel(const float* __restrict__ att,
                           const unsigned char* __restrict__ mask,
                           float* __restrict__ out) {
    int b = blockIdx.x, tid = threadIdx.x;
    __shared__ float sv[256];
    __shared__ int   si[256];

    float bestV = -INFINITY; int bestI = 0x7fffffff;
    for (int j = tid; j < 4 * SLEN; j += 256) {
        bool valid = (mask[b * SLEN + (j & (SLEN - 1))] == 0);
        float v = valid ? 10.0f * FAST_TANH(att[(size_t)b * 4 * SLEN + j])
                        : -INFINITY;
        if (v > bestV || (v == bestV && j < bestI)) { bestV = v; bestI = j; }
    }
    sv[tid] = bestV; si[tid] = bestI;
    __syncthreads();
    for (int s = 128; s > 0; s >>= 1) {
        if (tid < s) {
            if (sv[tid + s] > sv[tid] ||
                (sv[tid + s] == sv[tid] && si[tid + s] < si[tid])) {
                sv[tid] = sv[tid + s]; si[tid] = si[tid + s];
            }
        }
        __syncthreads();
    }
    float gmax = sv[0];
    int   gidx = si[0];
    __syncthreads();

    float sum = 0.0f;
    for (int j = tid; j < 4 * SLEN; j += 256) {
        bool valid = (mask[b * SLEN + (j & (SLEN - 1))] == 0);
        float v = valid ? 10.0f * FAST_TANH(att[(size_t)b * 4 * SLEN + j])
                        : -INFINITY;
        sum += expf(v - gmax);
    }
    sv[tid] = sum;
    __syncthreads();
    for (int s = 128; s > 0; s >>= 1) {
        if (tid < s) sv[tid] += sv[tid + s];
        __syncthreads();
    }
    if (tid == 0) {
        out[b]         = (float)gidx;      // indices[:,None]
        out[BATCH + b] = 1.0f / sv[0];     // p[:,None]
    }
    // mask passthrough
    for (int i = tid; i < SLEN; i += 256)
        out[2 * BATCH + b * SLEN + i] = mask[b * SLEN + i] ? 1.0f : 0.0f;
}

// ---------------------------------------------------------------------------
extern "C" void kernel_launch(void* const* d_in, const int* in_sizes, int n_in,
                              void* d_out, int out_size, void* d_ws, size_t ws_size,
                              hipStream_t stream) {
    const float* x       = (const float*)d_in[0];
    const float* context = (const float*)d_in[1];
    const unsigned char* mask = (const unsigned char*)d_in[2];
    const float* Wi  = (const float*)d_in[3];
    const float* bi  = (const float*)d_in[4];
    const float* Wc0 = (const float*)d_in[5];
    const float* bc0 = (const float*)d_in[6];
    const float* Wc1 = (const float*)d_in[7];
    const float* bc1 = (const float*)d_in[8];
    const float* Wc2 = (const float*)d_in[9];
    const float* bc2 = (const float*)d_in[10];
    const float* Wc3 = (const float*)d_in[11];
    const float* bc3 = (const float*)d_in[12];
    const float* V   = (const float*)d_in[13];

    char* ws = (char*)d_ws;
    __bf16* Bfrag = (__bf16*)ws;                                    // 2 MB
    float*  inp   = (float*)(ws + (size_t)4 * HDIM * DDIM * 2);     // 64 KB
    float*  att   = (float*)(ws + (size_t)4 * HDIM * DDIM * 2
                                + (size_t)BATCH * HDIM * 4);        // 1 MB

    prep_w    <<<256,  256, 0, stream>>>(Wc0, Wc1, Wc2, Wc3, Bfrag);
    inp_kernel<<<64,   256, 0, stream>>>(x, Wi, bi, inp);
    att_kernel<<<1024, 256, DYN_LDS_BYTES, stream>>>(context, Bfrag, inp,
                                                     bc0, bc1, bc2, bc3, V, att);
    out_kernel<<<BATCH, 256, 0, stream>>>(att, mask, (float*)d_out);
}
